// GRU_67714454389230
// MI455X (gfx1250) — matrix operand order repaired and checked by reading, not compiled, not run on hardware
//
#include <hip/hip_runtime.h>

// Fused 2-layer GRU + FC for MI455X (gfx1250, wave32, WMMA).
// One wave32 per workgroup owns a 16-row batch tile and runs the entire
// T=512 recurrence; all inter-layer state lives in registers/LDS.
// Biases are folded into the K-padding of the B operands (constant 1.0
// appended to the A inputs at K=12 / K=18), so every WMMA uses an inline-0
// C accumulator. x loads are software-pipelined one iteration ahead.

typedef __attribute__((ext_vector_type(16))) _Float16 v16h;
typedef __attribute__((ext_vector_type(8)))  float    v8f;

#define IN_DIM 18
#define H_DIM  12
#define B_DIM  4096
#define T_DIM  512
#define TILE_B 16
#define LOG2E  1.44269504088896340736f

static __device__ __forceinline__ v8f wmma_f16(v16h a, v16h b, v8f c) {
  // (neg_a, A, neg_b, B, c_mod, C, reuse_a, reuse_b)
  return __builtin_amdgcn_wmma_f32_16x16x32_f16(false, a, false, b, (short)0, c,
                                                false, false);
}

static __device__ __forceinline__ float fast_sigmoid(float x) {
  float e = __builtin_amdgcn_exp2f(-x * LOG2E);
  return __builtin_amdgcn_rcpf(1.0f + e);
}
static __device__ __forceinline__ float fast_tanh(float x) {
  float e = __builtin_amdgcn_exp2f(-2.0f * LOG2E * x);
  return (1.0f - e) * __builtin_amdgcn_rcpf(1.0f + e);
}

// B operand fragment for D = A(16xK) * B(Kx16). W is [3H, Din] row-major,
// gate g occupies rows g*12..g*12+11. B[k][n] = W[g*12+n][k] for k<Din,
// and B[Din][n] = bias0[n] (+ bias1[n]) -- the bias row, multiplied by the
// constant 1.0 planted at K=Din in the A operand.
// 16-bit B layout (wave32): lanes 0-15 hold K=0..7,16..23 in elements 0..15,
// lanes 16-31 hold K=8..15,24..31. Column = lane%16.
static __device__ __forceinline__ v16h make_bfrag(const float* __restrict__ W,
                                                  int Din, int gate,
                                                  const float* __restrict__ bias0,
                                                  const float* __restrict__ bias1,
                                                  int lane) {
  int col = lane & 15;
  bool hi = lane >= 16;
  v16h b;
  #pragma unroll
  for (int j = 0; j < 16; ++j) {
    int K = (j < 8) ? ((hi ? 8 : 0) + j) : ((hi ? 24 : 16) + (j - 8));
    float v = 0.0f;
    if (col < H_DIM) {
      if (K < Din) {
        v = W[(gate * H_DIM + col) * Din + K];
      } else if (K == Din) {  // bias row at K = Din (the 1.0 slot in A)
        if (bias0) v += bias0[gate * H_DIM + col];
        if (bias1) v += bias1[gate * H_DIM + col];
      }
    }
    b[j] = (_Float16)v;
  }
  return b;
}

// GRU elementwise update on accumulator fragments (pure per-lane VALU;
// r/z/n live in separate lane-aligned 16x16 tiles; biases already inside D).
static __device__ __forceinline__ v8f gru_update(v8f dir, v8f dhr, v8f diz,
                                                 v8f dhz, v8f din, v8f dhn,
                                                 v8f hold) {
  v8f h;
  #pragma unroll
  for (int v = 0; v < 8; ++v) {
    float r = fast_sigmoid(dir[v] + dhr[v]);
    float z = fast_sigmoid(diz[v] + dhz[v]);
    float n = fast_tanh(din[v] + r * dhn[v]);
    h[v] = (1.0f - z) * n + z * hold[v];
  }
  return h;
}

// Spill h (C/D fragment, f32) to LDS as f16 [16 rows][16 cols]. Columns
// 12..15 are never written: col 12 stays 1.0 (bias input), 13..15 stay 0.
static __device__ __forceinline__ void store_h(_Float16* buf, v8f h, int lane) {
  int col = lane & 15;
  int mb = (lane >= 16) ? 8 : 0;
  if (col < H_DIM) {
    #pragma unroll
    for (int v = 0; v < 8; ++v) buf[(mb + v) * 16 + col] = (_Float16)h[v];
  }
}

// Reload h from LDS in A-operand layout (16-bit A 16x32):
//   lo lanes: elements 0..7 = K 0..7   -> buf[row][0..7]
//   hi lanes: elements 0..7 = K 8..15  -> buf[row][8..15] (12 -> 1.0 bias)
// Elements 8..15 (K=16..31) are zero. One contiguous 16B LDS read per lane.
static __device__ __forceinline__ v16h load_hA(const _Float16* buf, int lane) {
  int row = lane & 15;
  int off = (lane >= 16) ? 8 : 0;
  v16h a;
  #pragma unroll
  for (int j = 0; j < 8; ++j) a[j] = buf[row * 16 + off + j];
  #pragma unroll
  for (int j = 8; j < 16; ++j) a[j] = (_Float16)0.0f;
  return a;
}

__global__ __launch_bounds__(32) void gru2_fused_wmma(
    const float* __restrict__ x,
    const float* __restrict__ w_ih0, const float* __restrict__ w_hh0,
    const float* __restrict__ b_ih0, const float* __restrict__ b_hh0,
    const float* __restrict__ w_ih1, const float* __restrict__ w_hh1,
    const float* __restrict__ b_ih1, const float* __restrict__ b_hh1,
    const float* __restrict__ fc_w,  const float* __restrict__ fc_b,
    float* __restrict__ out) {
  __shared__ _Float16 h1buf[TILE_B * 16];
  __shared__ _Float16 h2buf[TILE_B * 16];

  const int lane = (int)threadIdx.x;  // wave32 == workgroup
  const int col  = lane & 15;
  const bool hi  = lane >= 16;
  const int b0   = (int)blockIdx.x * TILE_B;

  // Init LDS h buffers: zeros except column 12 == 1.0 (the bias-row input).
  for (int i = lane; i < TILE_B * 16; i += 32) {
    _Float16 v = ((i & 15) == H_DIM) ? (_Float16)1.0f : (_Float16)0.0f;
    h1buf[i] = v;
    h2buf[i] = v;
  }
  __syncthreads();

  // ---- Constant B operand fragments (held in VGPRs for all of T) ----
  // gi biases: r/z carry b_ih+b_hh (they sum linearly inside sigmoid);
  // n carries b_ih only (b_hh_n goes through the gh path inside r*(.)).
  v16h bW0r = make_bfrag(w_ih0, IN_DIM, 0, b_ih0, b_hh0, lane);
  v16h bW0z = make_bfrag(w_ih0, IN_DIM, 1, b_ih0, b_hh0, lane);
  v16h bW0n = make_bfrag(w_ih0, IN_DIM, 2, b_ih0, nullptr, lane);
  v16h bU0r = make_bfrag(w_hh0, H_DIM, 0, nullptr, nullptr, lane);
  v16h bU0z = make_bfrag(w_hh0, H_DIM, 1, nullptr, nullptr, lane);
  v16h bU0n = make_bfrag(w_hh0, H_DIM, 2, b_hh0, nullptr, lane);
  v16h bW1r = make_bfrag(w_ih1, H_DIM, 0, b_ih1, b_hh1, lane);
  v16h bW1z = make_bfrag(w_ih1, H_DIM, 1, b_ih1, b_hh1, lane);
  v16h bW1n = make_bfrag(w_ih1, H_DIM, 2, b_ih1, nullptr, lane);
  v16h bU1r = make_bfrag(w_hh1, H_DIM, 0, nullptr, nullptr, lane);
  v16h bU1z = make_bfrag(w_hh1, H_DIM, 1, nullptr, nullptr, lane);
  v16h bU1n = make_bfrag(w_hh1, H_DIM, 2, b_hh1, nullptr, lane);

  // FC as one more WMMA: only column 0 of B is non-zero (fc_w; fc_b at K=12).
  v16h bFC;
  #pragma unroll
  for (int j = 0; j < 16; ++j) {
    int K = (j < 8) ? ((hi ? 8 : 0) + j) : ((hi ? 24 : 16) + (j - 8));
    float v = 0.0f;
    if (col == 0) {
      if (K < H_DIM) v = fc_w[K];
      else if (K == H_DIM) v = fc_b[0];
    }
    bFC[j] = (_Float16)v;
  }

  v8f czero;
  #pragma unroll
  for (int i = 0; i < 8; ++i) czero[i] = 0.0f;

  // ---- Recurrent state (h0 == 0; 1.0 planted at K=12 on hi lanes) ----
  v8f  h1 = czero, h2 = czero;  // f32, WMMA C/D layout
  v16h h1A, h2A;                // f16, WMMA A layout
  #pragma unroll
  for (int j = 0; j < 16; ++j) { h1A[j] = (_Float16)0.0f; h2A[j] = (_Float16)0.0f; }
  if (hi) { h1A[4] = (_Float16)1.0f; h2A[4] = (_Float16)1.0f; }  // K=12

  // x addressing: lo lanes read K=0..7 from row offset 0, hi lanes K=8..15
  // from +8. Tail (K=16,17) is loaded uniformly by ALL lanes from +16 of the
  // same row (branchless, in-bounds) and selected into lo lanes only.
  const int row = col;
  const size_t rbase = (size_t)(b0 + row) * (size_t)T_DIM * IN_DIM;
  const float* xmain = x + rbase + (hi ? 8 : 0);
  const float* xtail = x + rbase + 16;

  // Software pipeline: raw f32 x values for the *current* step in registers,
  // loads for the next step issued at the top of each iteration.
  float cur[10], nxt[10];
  #pragma unroll
  for (int j = 0; j < 8; ++j) cur[j] = xmain[j];
  cur[8] = xtail[0];
  cur[9] = xtail[1];

  for (int t = 0; t < T_DIM; ++t) {
    // Issue next step's loads now; they retire behind this iteration's math.
    {
      int tn = (t + 1 < T_DIM) ? (t + 1) : t;  // clamp: stay in bounds
      const float* xm = xmain + (size_t)tn * IN_DIM;
      const float* xt = xtail + (size_t)tn * IN_DIM;
      #pragma unroll
      for (int j = 0; j < 8; ++j) nxt[j] = xm[j];
      nxt[8] = xt[0];
      nxt[9] = xt[1];
    }

    // Build the x_t A fragment from the prefetched raw values (branchless).
    v16h xA;
    #pragma unroll
    for (int j = 0; j < 8; ++j) xA[j] = (_Float16)cur[j];  // K=0..7 / K=8..15
    #pragma unroll
    for (int j = 11; j < 16; ++j) xA[j] = (_Float16)0.0f;
    xA[8]  = hi ? (_Float16)0.0f : (_Float16)cur[8];  // K=16
    xA[9]  = hi ? (_Float16)0.0f : (_Float16)cur[9];  // K=17
    xA[10] = hi ? (_Float16)0.0f : (_Float16)1.0f;    // K=18: bias row input

    // ---- Layer 0 ----
    v8f dir = wmma_f16(xA,  bW0r, czero);
    v8f diz = wmma_f16(xA,  bW0z, czero);
    v8f din = wmma_f16(xA,  bW0n, czero);
    v8f dhr = wmma_f16(h1A, bU0r, czero);
    v8f dhz = wmma_f16(h1A, bU0z, czero);
    v8f dhn = wmma_f16(h1A, bU0n, czero);
    h1 = gru_update(dir, dhr, diz, dhz, din, dhn, h1);

    store_h(h1buf, h1, lane);
    __syncthreads();
    h1A = load_hA(h1buf, lane);  // used for gi1 now, gh0 at t+1

    // ---- Layer 1 ----
    dir = wmma_f16(h1A, bW1r, czero);
    diz = wmma_f16(h1A, bW1z, czero);
    din = wmma_f16(h1A, bW1n, czero);
    dhr = wmma_f16(h2A, bU1r, czero);
    dhz = wmma_f16(h2A, bU1z, czero);
    dhn = wmma_f16(h2A, bU1n, czero);
    h2 = gru_update(dir, dhr, diz, dhz, din, dhn, h2);

    store_h(h2buf, h2, lane);
    __syncthreads();
    h2A = load_hA(h2buf, lane);  // used for FC now, gh1 at t+1

    // ---- FC: out[:, t] = h2 . fc_w + fc_b (column 0 of D) ----
    v8f o = wmma_f16(h2A, bFC, czero);
    if (col == 0) {
      int mb = hi ? 8 : 0;
      #pragma unroll
      for (int v = 0; v < 8; ++v)
        out[(size_t)(b0 + mb + v) * T_DIM + t] = o[v];
    }

    // Rotate the x pipeline.
    #pragma unroll
    for (int j = 0; j < 10; ++j) cur[j] = nxt[j];
  }
}

extern "C" void kernel_launch(void* const* d_in, const int* in_sizes, int n_in,
                              void* d_out, int out_size, void* d_ws,
                              size_t ws_size, hipStream_t stream) {
  (void)in_sizes; (void)n_in; (void)d_ws; (void)ws_size; (void)out_size;
  const float* x     = (const float*)d_in[0];
  const float* w_ih0 = (const float*)d_in[1];
  const float* w_hh0 = (const float*)d_in[2];
  const float* b_ih0 = (const float*)d_in[3];
  const float* b_hh0 = (const float*)d_in[4];
  const float* w_ih1 = (const float*)d_in[5];
  const float* w_hh1 = (const float*)d_in[6];
  const float* b_ih1 = (const float*)d_in[7];
  const float* b_hh1 = (const float*)d_in[8];
  const float* fc_w  = (const float*)d_in[9];
  const float* fc_b  = (const float*)d_in[10];
  float* out = (float*)d_out;

  dim3 grid(B_DIM / TILE_B);  // 256 independent wave32 workgroups
  dim3 block(32);
  gru2_fused_wmma<<<grid, block, 0, stream>>>(x, w_ih0, w_hh0, b_ih0, b_hh0,
                                              w_ih1, w_hh1, b_ih1, b_hh1,
                                              fc_w, fc_b, out);
}
